// HeteroRGCN_42803644072648
// MI455X (gfx1250) — matrix operand order repaired and checked by reading, not compiled, not run on hardware
//
#include <hip/hip_runtime.h>
#include <hip/hip_bf16.h>
#include <stdint.h>

#define H 256
#define RNUM 5

typedef __attribute__((ext_vector_type(16))) __bf16 v16bf;
typedef __attribute__((ext_vector_type(8)))  float  v8f;

struct alignas(16) U4 { unsigned x, y, z, w; };
union FragBF { v16bf v; U4 u[2]; };

__device__ __forceinline__ unsigned short f2bf(float f) {
    union { float f; unsigned u; } x; x.f = f;
    unsigned r = x.u + 0x7FFFu + ((x.u >> 16) & 1u);   // round-to-nearest-even
    return (unsigned short)(r >> 16);
}

// async copy 16 bytes: global -> LDS (ASYNCcnt-tracked, CDNA5 path)
__device__ __forceinline__ void async_g2l_b128(unsigned lds_byte_addr,
                                               unsigned long long gaddr) {
    asm volatile("global_load_async_to_lds_b128 %0, %1, off"
                 :: "v"(lds_byte_addr), "v"(gaddr) : "memory");
}
__device__ __forceinline__ void wait_async0() {
    asm volatile("s_wait_asynccnt 0" ::: "memory");
}

// ---------------------------------------------------------------- utilities
__global__ void fill_i32(int* p, int n, int v) {
    int i = blockIdx.x * blockDim.x + threadIdx.x;
    if (i < n) p[i] = v;
}
__global__ void fill_f32(float* p, long long n, float v) {
    long long i = (long long)blockIdx.x * blockDim.x + threadIdx.x;
    if (i < n) p[i] = v;
}
// f32 -> bf16 elementwise (initial activation conversion)
__global__ void convert_rows(const float* __restrict__ in,
                             unsigned short* __restrict__ out, long long n) {
    long long i = (long long)blockIdx.x * blockDim.x + threadIdx.x;
    if (i < n) out[i] = f2bf(in[i]);
}
// lrelu(acc) -> bf16 act buffer and/or f32 output
__global__ void finalize_k(const float* __restrict__ acc,
                           unsigned short* __restrict__ obf,
                           float* __restrict__ of32, long long n) {
    long long i = (long long)blockIdx.x * blockDim.x + threadIdx.x;
    if (i < n) {
        float v = acc[i];
        v = v > 0.0f ? v : 0.01f * v;
        if (obf)  obf[i]  = f2bf(v);
        if (of32) of32[i] = v;
    }
}

// ----------------------------------------------------------- edge binning
// meta layout: [0..4]=hist, [5..10]=off(6, 16-aligned), [11..15]=cursor
__global__ void hist_et(const int* __restrict__ et, int E, int* __restrict__ meta) {
    int i = blockIdx.x * blockDim.x + threadIdx.x;
    if (i < E) atomicAdd(&meta[et[i]], 1);
}
__global__ void scan_bins(int* meta) {
    if (threadIdx.x == 0 && blockIdx.x == 0) {
        int acc = 0;
        for (int r = 0; r < RNUM; ++r) {
            meta[5 + r] = acc;
            acc += (meta[r] + 15) & ~15;
            meta[11 + r] = 0;
        }
        meta[5 + RNUM] = acc;
    }
}
__global__ void scatter_edges(const int* __restrict__ et, int E,
                              int* __restrict__ meta, int* __restrict__ list) {
    int i = blockIdx.x * blockDim.x + threadIdx.x;
    if (i < E) {
        int r = et[i];
        int pos = atomicAdd(&meta[11 + r], 1);
        list[meta[5 + r] + pos] = i;
    }
}
__global__ void count_edges(const int* __restrict__ et, const int* __restrict__ dstI,
                            int E, float* __restrict__ cnt, int n_dst) {
    int i = blockIdx.x * blockDim.x + threadIdx.x;
    if (i < E) atomicAdd(&cnt[(long long)et[i] * n_dst + dstI[i]], 1.0f);
}
__global__ void invert_cnt(float* cnt, long long n) {
    long long i = (long long)blockIdx.x * blockDim.x + threadIdx.x;
    if (i < n) cnt[i] = 1.0f / fmaxf(cnt[i], 1.0f);
}

// --------------------------------------- weight convert: f32 [k][n] -> bf16 [n][k]
__global__ void convert_w(const float* __restrict__ in, unsigned short* __restrict__ outp,
                          int n_mats) {
    long long i = (long long)blockIdx.x * blockDim.x + threadIdx.x;
    long long total = (long long)n_mats * H * H;
    if (i >= total) return;
    long long mat = i >> 16;
    int rem = (int)(i & 65535);
    int k = rem >> 8, n = rem & 255;
    outp[(mat << 16) + (long long)n * H + k] = f2bf(in[i]);
}

// ------------------------------------------------------------ root GEMM
// acc[row,:] = act_bf16[row,:] @ root^T(bf16 [n][k]) + bias ; 16 rows per block
__global__ __launch_bounds__(512) void root_gemm(
    const unsigned short* __restrict__ act, const unsigned short* __restrict__ wt,
    const float* __restrict__ bias, float* __restrict__ acc_out, int n) {
    __shared__ __align__(16) unsigned short ldsA[16 * H];
    const int tid = threadIdx.x;
    const int tile_m = blockIdx.x;
    {   // async-stage 16 bf16 rows (8 KB) into LDS: one b128 per lane
        int m = tid >> 5;
        int chunk = (tid & 31) * 16;                 // bytes within 512B row
        int row = tile_m * 16 + m;
        if (row >= n) row = n - 1;                   // clamp; masked at store
        unsigned long long ga = (unsigned long long)(uintptr_t)act
                              + (unsigned long long)row * (H * 2) + chunk;
        unsigned la = (unsigned)(uintptr_t)(&ldsA[0]) + (unsigned)(tid * 16);
        async_g2l_b128(la, ga);
    }
    wait_async0();
    __syncthreads();
    const int wave = tid >> 5;       // 16 waves -> 16 column tiles
    const int lane = tid & 31;
    const int half = lane >> 4;
    const int lidx = lane & 15;
    v8f acc;
    #pragma unroll
    for (int i = 0; i < 8; ++i) acc[i] = 0.0f;
    const unsigned short* wrow = wt + (long long)(wave * 16 + lidx) * H;
    #pragma unroll
    for (int kb = 0; kb < 8; ++kb) {
        FragBF a, b;
        int ka = kb * 32 + half * 8;          // A: K chunks at kbase, kbase+16
        a.u[0] = *(const U4*)&ldsA[lidx * H + ka];
        a.u[1] = *(const U4*)&ldsA[lidx * H + ka + 16];
        int kB = kb * 32 + half * 8;          // B: K chunks at kbase, kbase+16
        b.u[0] = *(const U4*)&wrow[kB];
        b.u[1] = *(const U4*)&wrow[kB + 16];
        acc = __builtin_amdgcn_wmma_f32_16x16x32_bf16(false, a.v, false, b.v,
                                                      (short)0, acc, false, false);
    }
    int col = wave * 16 + lidx;
    float bv = bias[col];
    #pragma unroll
    for (int i = 0; i < 8; ++i) {
        int row = tile_m * 16 + half * 8 + i;
        if (row < n) acc_out[(long long)row * H + col] = acc[i] + bv;
    }
}

// ------------------------------------------------------------ edge GEMM
// 16 edges (single relation, padded bin list) x W_r, scaled scatter-add into acc
__global__ __launch_bounds__(512) void edge_gemm(
    const unsigned short* __restrict__ xsrc, const unsigned short* __restrict__ wt5,
    const float* __restrict__ invc, const int* __restrict__ srcI,
    const int* __restrict__ dstI, const int* __restrict__ list,
    const int* __restrict__ meta, float* __restrict__ acc_out, int n_dst) {
    __shared__ __align__(16) unsigned short ldsA[16 * H];
    __shared__ int   s_src[16];
    __shared__ int   s_dst[16];
    __shared__ float s_scl[16];
    __shared__ int   s_r;
    const int tid  = threadIdx.x;
    const int base = blockIdx.x * 16;
    const int* off = meta + 5;
    if (base >= off[RNUM]) return;
    if (tid == 0) {
        int r = 0;
        while (r < RNUM - 1 && base >= off[r + 1]) ++r;
        s_r = r;
    }
    if (tid < 16) {
        int eid = list[base + tid];
        if (eid >= 0) { s_src[tid] = srcI[eid]; s_dst[tid] = dstI[eid]; }
        else          { s_src[tid] = -1;        s_dst[tid] = -1;        }
        s_scl[tid] = 0.0f;
    }
    __syncthreads();
    const int r = s_r;
    {   // async-gather 16 scattered bf16 rows into LDS
        int m = tid >> 5;
        int chunk = (tid & 31) * 16;
        int sn = s_src[m];
        if (sn < 0) sn = 0;                          // masked at store
        unsigned long long ga = (unsigned long long)(uintptr_t)xsrc
                              + (unsigned long long)sn * (H * 2) + chunk;
        unsigned la = (unsigned)(uintptr_t)(&ldsA[0]) + (unsigned)(tid * 16);
        async_g2l_b128(la, ga);
    }
    if (tid < 16 && s_dst[tid] >= 0)
        s_scl[tid] = invc[(long long)r * n_dst + s_dst[tid]];
    wait_async0();
    __syncthreads();
    const int wave = tid >> 5;
    const int lane = tid & 31;
    const int half = lane >> 4;
    const int lidx = lane & 15;
    v8f acc;
    #pragma unroll
    for (int i = 0; i < 8; ++i) acc[i] = 0.0f;
    const unsigned short* wrow = wt5 + (long long)r * H * H
                               + (long long)(wave * 16 + lidx) * H;
    #pragma unroll
    for (int kb = 0; kb < 8; ++kb) {
        FragBF a, b;
        int ka = kb * 32 + half * 8;
        a.u[0] = *(const U4*)&ldsA[lidx * H + ka];
        a.u[1] = *(const U4*)&ldsA[lidx * H + ka + 16];
        int kB = kb * 32 + half * 8;
        b.u[0] = *(const U4*)&wrow[kB];
        b.u[1] = *(const U4*)&wrow[kB + 16];
        acc = __builtin_amdgcn_wmma_f32_16x16x32_bf16(false, a.v, false, b.v,
                                                      (short)0, acc, false, false);
    }
    int col = wave * 16 + lidx;
    #pragma unroll
    for (int i = 0; i < 8; ++i) {
        int ml = half * 8 + i;
        int d  = s_dst[ml];
        if (d >= 0) atomicAdd(&acc_out[(long long)d * H + col], acc[i] * s_scl[ml]);
    }
}

// =========================================================== host driver
extern "C" void kernel_launch(void* const* d_in, const int* in_sizes, int n_in,
                              void* d_out, int out_size, void* d_ws, size_t ws_size,
                              hipStream_t stream) {
    (void)n_in; (void)out_size; (void)ws_size;
    const float* xin[5] = {(const float*)d_in[0], (const float*)d_in[1],
                           (const float*)d_in[2], (const float*)d_in[3],
                           (const float*)d_in[4]};
    const float* W    = (const float*)d_in[5];
    const float* root = (const float*)d_in[6];
    const float* bias = (const float*)d_in[7];
    const int* srcE[3] = {(const int*)d_in[8],  (const int*)d_in[11], (const int*)d_in[14]};
    const int* dstE[3] = {(const int*)d_in[9],  (const int*)d_in[12], (const int*)d_in[15]};
    const int* etE[3]  = {(const int*)d_in[10], (const int*)d_in[13], (const int*)d_in[16]};
    const int E[3] = {in_sizes[8], in_sizes[11], in_sizes[14]};
    const int nNode[5] = {in_sizes[0] / H, in_sizes[1] / H, in_sizes[2] / H,
                          in_sizes[3] / H, in_sizes[4] / H};
    const int nm = nNode[0], nf = nNode[3], nl = nNode[4];
    const int nDst[3] = {nm, nf, nl};

    char* ws = (char*)d_ws;
    size_t wo = 0;
    auto alloc = [&](size_t bytes) -> void* {
        void* p = (void*)(ws + wo);
        wo += (bytes + 255) & ~(size_t)255;
        return p;
    };
    float* accBuf = (float*)alloc(sizeof(float) * (size_t)(nm + nf + nl) * H);
    unsigned short* actBf[5];
    for (int t = 0; t < 5; ++t)
        actBf[t] = (unsigned short*)alloc(sizeof(unsigned short) * (size_t)nNode[t] * H);
    unsigned short* Wt    = (unsigned short*)alloc(sizeof(unsigned short) * 45ull * H * H);
    unsigned short* rootT = (unsigned short*)alloc(sizeof(unsigned short) * 9ull  * H * H);
    float* invc[3]; int* lst[3]; int* meta[3];
    for (int t = 0; t < 3; ++t) invc[t] = (float*)alloc(sizeof(float) * (size_t)RNUM * nDst[t]);
    for (int t = 0; t < 3; ++t) lst[t]  = (int*)alloc(sizeof(int) * (size_t)(E[t] + RNUM * 16));
    for (int t = 0; t < 3; ++t) meta[t] = (int*)alloc(sizeof(int) * 16);

    // ---- per-call prep (edges & weights fixed across layers) ----
    for (int t = 0; t < 5; ++t) {
        long long ne = (long long)nNode[t] * H;
        convert_rows<<<(unsigned)((ne + 255) / 256), 256, 0, stream>>>(xin[t], actBf[t], ne);
    }
    for (int t = 0; t < 3; ++t) {
        fill_i32<<<1, 64, 0, stream>>>(meta[t], 16, 0);
        long long nc = (long long)RNUM * nDst[t];
        fill_f32<<<(unsigned)((nc + 255) / 256), 256, 0, stream>>>(invc[t], nc, 0.0f);
        fill_i32<<<(E[t] + RNUM * 16 + 255) / 256, 256, 0, stream>>>(lst[t], E[t] + RNUM * 16, -1);
    }
    for (int t = 0; t < 3; ++t)
        hist_et<<<(E[t] + 255) / 256, 256, 0, stream>>>(etE[t], E[t], meta[t]);
    for (int t = 0; t < 3; ++t)
        scan_bins<<<1, 1, 0, stream>>>(meta[t]);
    for (int t = 0; t < 3; ++t)
        scatter_edges<<<(E[t] + 255) / 256, 256, 0, stream>>>(etE[t], E[t], meta[t], lst[t]);
    for (int t = 0; t < 3; ++t) {
        count_edges<<<(E[t] + 255) / 256, 256, 0, stream>>>(etE[t], dstE[t], E[t], invc[t], nDst[t]);
        long long nc = (long long)RNUM * nDst[t];
        invert_cnt<<<(unsigned)((nc + 255) / 256), 256, 0, stream>>>(invc[t], nc);
    }
    convert_w<<<(45 * 65536 + 255) / 256, 256, 0, stream>>>(W, Wt, 45);
    convert_w<<<( 9 * 65536 + 255) / 256, 256, 0, stream>>>(root, rootT, 9);

    // ---- 3 layers; activations live as bf16, acc is f32 ----
    float* dm = (float*)d_out;
    float* df = dm + (size_t)nm * H;
    float* dl = df + (size_t)nf * H;
    float* accP[3] = {accBuf, accBuf + (size_t)nm * H, accBuf + (size_t)(nm + nf) * H};
    unsigned short* aSrc[3] = {actBf[0], actBf[1], actBf[2]};  // miner, validator, storage
    unsigned short* aDst[3] = {actBf[0], actBf[3], actBf[4]};  // miner, full, light
    float* outF32[3] = {dm, df, dl};

    for (int L = 0; L < 3; ++L) {
        for (int c = 0; c < 3; ++c) {
            const int n = nDst[c];
            const unsigned short* wr = rootT + (size_t)(L * 3 + c) * H * H;
            const unsigned short* we = Wt    + (size_t)(L * 3 + c) * RNUM * H * H;
            const float*          bb = bias  + (size_t)(L * 3 + c) * H;
            root_gemm<<<(n + 15) / 16, 512, 0, stream>>>(aDst[c], wr, bb, accP[c], n);
            edge_gemm<<<(E[c] + 15) / 16 + RNUM, 512, 0, stream>>>(
                aSrc[c], we, invc[c], srcE[c], dstE[c], lst[c], meta[c], accP[c], n);
            long long tot = (long long)n * H;
            finalize_k<<<(unsigned)((tot + 255) / 256), 256, 0, stream>>>(
                accP[c], (L < 2) ? aDst[c] : (unsigned short*)nullptr,
                (L == 2) ? outF32[c] : (float*)nullptr, tot);
        }
    }
}
